// FineGrainedMoE_17927193493784
// MI455X (gfx1250) — matrix operand order repaired and checked by reading, not compiled, not run on hardware
//
#include <hip/hip_runtime.h>

#define TT   4096      // tokens
#define HH   1024      // hidden
#define FF   1024      // ffn
#define EE   64        // experts
#define CAP  256       // capacity per expert
#define TKP  8192      // T * K pairs

typedef __bf16 bf16;
typedef bf16  v16bf __attribute__((ext_vector_type(16)));
typedef bf16  v8bf  __attribute__((ext_vector_type(8)));
typedef float v8f   __attribute__((ext_vector_type(8)));

union FragAB { v16bf v; v8bf h[2]; };

// gfx1250 async global->LDS path via inline asm (ASYNCcnt-tracked LDS DMA)
#if defined(__gfx1250__)
#define USE_ASYNC_LDS 1
#else
#define USE_ASYNC_LDS 0
#endif

// ---------------- init ----------------
__global__ void init_counters(int* counters) {
    if (threadIdx.x < EE) counters[threadIdx.x] = 0;
}

// ---------------- router: logits, softmax, top-2, dispatch ----------------
__global__ __launch_bounds__(64) void router_kernel(
    const float* __restrict__ x, const float* __restrict__ gate_w,
    float* __restrict__ probs, float* __restrict__ topw,
    int* __restrict__ dest, int* __restrict__ counters,
    bf16* __restrict__ xb)
{
    const int t = blockIdx.x;
    const int e = threadIdx.x;
    __shared__ float xs[HH];
    __shared__ float pr[EE];
    __shared__ float sred[2];
    __shared__ int   sd[2];

    const float* xrow = x + (size_t)t * HH;
    #pragma unroll
    for (int j = 0; j < HH / 64; ++j) xs[e + j * 64] = xrow[e + j * 64];
    __syncthreads();

    const float* gw = gate_w + (size_t)e * HH;
    float acc = 0.f;
    for (int h = 0; h < HH; h += 4)
        acc += gw[h] * xs[h] + gw[h+1] * xs[h+1] + gw[h+2] * xs[h+2] + gw[h+3] * xs[h+3];
    pr[e] = acc;
    __syncthreads();

    if (e == 0) { float mx = pr[0]; for (int i = 1; i < EE; ++i) mx = fmaxf(mx, pr[i]); sred[0] = mx; }
    __syncthreads();
    float p = __expf(acc - sred[0]);
    pr[e] = p;
    __syncthreads();
    if (e == 0) { float s = 0.f; for (int i = 0; i < EE; ++i) s += pr[i]; sred[1] = s; }
    __syncthreads();
    p = p / sred[1];
    probs[(size_t)t * EE + e] = p;
    pr[e] = p;
    __syncthreads();

    if (e == 0) {
        int i1 = 0; float p1 = pr[0];
        for (int i = 1; i < EE; ++i) if (pr[i] > p1) { p1 = pr[i]; i1 = i; }
        int i2 = (i1 == 0) ? 1 : 0; float p2 = -1.f;
        for (int i = 0; i < EE; ++i) if (i != i1 && pr[i] > p2) { p2 = pr[i]; i2 = i; }
        float inv = 1.f / (p1 + p2 + 1e-9f);
        topw[2 * t]     = p1 * inv;
        topw[2 * t + 1] = p2 * inv;
        int s1 = atomicAdd(&counters[i1], 1);
        int s2 = atomicAdd(&counters[i2], 1);
        int d0 = (s1 < CAP) ? i1 * CAP + s1 : -1;
        int d1 = (s2 < CAP) ? i2 * CAP + s2 : -1;
        dest[2 * t] = d0; dest[2 * t + 1] = d1;
        sd[0] = d0; sd[1] = d1;
    }
    __syncthreads();
    #pragma unroll
    for (int k = 0; k < 2; ++k) {
        int d = sd[k];
        if (d >= 0) {
            bf16* drow = xb + (size_t)d * HH;
            #pragma unroll
            for (int j = 0; j < HH / 64; ++j) drow[e + j * 64] = (bf16)xs[e + j * 64];
        }
    }
}

// ---------------- expert GEMM (bf16 WMMA, f32 accum) ----------------
#define BM 128
#define BN 128
#define BK 32
#define BSTR (BK + 8)   // padded LDS stride (40 bf16 = 80B; rows spread over banks, 16B aligned)

template <bool DUAL>
__global__ __launch_bounds__(256) void expert_gemm(
    const bf16*  __restrict__ A,     // [E][CAP][1024] bf16
    const float* __restrict__ B0,    // [E][1024][1024] f32 (w_gate or w_down)
    const float* __restrict__ B1,    // [E][1024][1024] f32 (w_up) when DUAL
    bf16*  __restrict__ Hout,        // DUAL: silu(g)*u as bf16
    float* __restrict__ Yout)        // !DUAL: y as f32
{
    const int e   = blockIdx.z;
    const int n0  = blockIdx.x * BN;
    const int m0  = blockIdx.y * BM;
    const int tid = threadIdx.x;
    const int wave = tid >> 5, lane = tid & 31;
    const int wm = wave >> 1, wn = wave & 1;          // 4x2 wave grid -> wave tile 32x64
    const int laneM = lane & 15, laneH = lane >> 4;

    __shared__ __align__(16) bf16 As [BM][BSTR];
    __shared__ __align__(16) bf16 Bs0[BN][BSTR];
    __shared__ __align__(16) bf16 Bs1[DUAL ? BN : 1][BSTR];

    v8f acc0[2][4] = {};
    v8f acc1[2][4] = {};

    const bf16*  Ag  = A  + ((size_t)e * CAP + m0) * (size_t)HH;
    const float* Bg0 = B0 + (size_t)e * HH * FF + n0;
    const float* Bg1 = DUAL ? (B1 + (size_t)e * HH * FF + n0) : (const float*)0;

    for (int kk = 0; kk < HH; kk += BK) {
        // prefetch next K-step's fp32 weight tile (global_prefetch_b8): 256 x 64B = 32x128 f32
        if (kk + BK < HH) {
            int prow = tid >> 3;               // 0..31
            int pcol = (tid & 7) * 16;         // 0..112 (x4B = 64B granules)
            __builtin_prefetch(Bg0 + (size_t)(kk + BK + prow) * FF + pcol, 0, 0);
            if (DUAL)
                __builtin_prefetch(Bg1 + (size_t)(kk + BK + prow) * FF + pcol, 0, 0);
        }

        // A tile: 128x32 bf16 (already bf16 in memory -> direct async LDS DMA on gfx1250)
#if USE_ASYNC_LDS
        #pragma unroll
        for (int j = 0; j < 2; ++j) {
            int idx = tid + j * 256;           // 0..511 x 16B chunks
            int row = idx >> 2;
            int c8  = (idx & 3) * 8;
            unsigned long long ga = (unsigned long long)(size_t)(Ag + (size_t)row * HH + kk + c8);
            unsigned ldsOff = (unsigned)(size_t)&As[row][c8];   // generic LDS addr[31:0] = LDS offset
            asm volatile("global_load_async_to_lds_b128 %0, %1, off"
                         :: "v"(ldsOff), "v"(ga) : "memory");
        }
#else
        #pragma unroll
        for (int j = 0; j < 4; ++j) {
            int idx = tid + j * 256;           // 0..1023
            int row = idx >> 3;
            int c4  = (idx & 7) * 4;
            uint2 v = *(const uint2*)(Ag + (size_t)row * HH + kk + c4);
            *(uint2*)&As[row][c4] = v;
        }
#endif
        // B tiles: 32x128 f32 -> transpose to N-major bf16 in LDS (conversion forces VGPR path)
        #pragma unroll
        for (int j = 0; j < 4; ++j) {
            int idx = tid + j * 256;           // 0..1023
            int bk  = idx >> 5;                // 0..31
            int n4  = (idx & 31) * 4;          // 0..124
            float4 w0 = *(const float4*)(Bg0 + (size_t)(kk + bk) * FF + n4);
            Bs0[n4 + 0][bk] = (bf16)w0.x;
            Bs0[n4 + 1][bk] = (bf16)w0.y;
            Bs0[n4 + 2][bk] = (bf16)w0.z;
            Bs0[n4 + 3][bk] = (bf16)w0.w;
            if (DUAL) {
                float4 w1 = *(const float4*)(Bg1 + (size_t)(kk + bk) * FF + n4);
                Bs1[n4 + 0][bk] = (bf16)w1.x;
                Bs1[n4 + 1][bk] = (bf16)w1.y;
                Bs1[n4 + 2][bk] = (bf16)w1.z;
                Bs1[n4 + 3][bk] = (bf16)w1.w;
            }
        }
#if USE_ASYNC_LDS
        asm volatile("s_wait_asynccnt 0x0" ::: "memory");   // our async LDS writes landed
#endif
        __syncthreads();

        // A fragments: 16x32 bf16 layout -> lane<16: K{0..7,16..23}, lane>=16: K{8..15,24..31}
        FragAB a[2];
        #pragma unroll
        for (int mi = 0; mi < 2; ++mi) {
            int row = wm * 32 + mi * 16 + laneM;
            int ko  = laneH * 8;
            a[mi].h[0] = *(const v8bf*)&As[row][ko];
            a[mi].h[1] = *(const v8bf*)&As[row][ko + 16];
        }
        // B fragments: 32x16 layout -> lane's column, lane<16: K{0..7,16..23}, lane>=16: K{8..15,24..31} (N-major LDS)
        #pragma unroll
        for (int ni = 0; ni < 4; ++ni) {
            int ncol = wn * 64 + ni * 16 + laneM;
            int ko   = laneH * 8;
            FragAB b0;
            b0.h[0] = *(const v8bf*)&Bs0[ncol][ko];
            b0.h[1] = *(const v8bf*)&Bs0[ncol][ko + 16];
            #pragma unroll
            for (int mi = 0; mi < 2; ++mi)
                acc0[mi][ni] = __builtin_amdgcn_wmma_f32_16x16x32_bf16(
                    false, a[mi].v, false, b0.v, (short)0, acc0[mi][ni], false, false);
            if (DUAL) {
                FragAB b1;
                b1.h[0] = *(const v8bf*)&Bs1[ncol][ko];
                b1.h[1] = *(const v8bf*)&Bs1[ncol][ko + 16];
                #pragma unroll
                for (int mi = 0; mi < 2; ++mi)
                    acc1[mi][ni] = __builtin_amdgcn_wmma_f32_16x16x32_bf16(
                        false, a[mi].v, false, b1.v, (short)0, acc1[mi][ni], false, false);
            }
        }
        __syncthreads();
    }

    // Epilogue. C/D layout: VGPR r -> M = r + (lane>=16 ? 8 : 0), N = lane%16
    #pragma unroll
    for (int mi = 0; mi < 2; ++mi) {
        #pragma unroll
        for (int ni = 0; ni < 4; ++ni) {
            #pragma unroll
            for (int r = 0; r < 8; ++r) {
                int row = m0 + wm * 32 + mi * 16 + laneH * 8 + r;
                int col = n0 + wn * 64 + ni * 16 + laneM;
                size_t off = ((size_t)e * CAP + row) * (size_t)FF + col;
                if (DUAL) {
                    float g = acc0[mi][ni][r];
                    float u = acc1[mi][ni][r];
                    // silu(g)*u with v_rcp_f32 instead of IEEE divide
                    float s = __builtin_amdgcn_rcpf(1.f + __expf(-g));
                    Hout[off] = (bf16)(g * s * u);
                } else {
                    Yout[off] = acc0[mi][ni][r];
                }
            }
        }
    }
}

// ---------------- combine ----------------
__global__ __launch_bounds__(256) void combine_kernel(
    const int* __restrict__ dest, const float* __restrict__ topw,
    const float* __restrict__ y, float* __restrict__ out)
{
    int t  = blockIdx.x;
    int d0 = dest[2 * t], d1 = dest[2 * t + 1];
    float w0 = topw[2 * t], w1 = topw[2 * t + 1];
    int idx = threadIdx.x * 4;
    float4 r = make_float4(0.f, 0.f, 0.f, 0.f);
    if (d0 >= 0) {
        float4 a = *(const float4*)(y + (size_t)d0 * HH + idx);
        r.x += w0 * a.x; r.y += w0 * a.y; r.z += w0 * a.z; r.w += w0 * a.w;
    }
    if (d1 >= 0) {
        float4 a = *(const float4*)(y + (size_t)d1 * HH + idx);
        r.x += w1 * a.x; r.y += w1 * a.y; r.z += w1 * a.z; r.w += w1 * a.w;
    }
    *(float4*)(out + (size_t)t * HH + idx) = r;
}

// ---------------- aux loss ----------------
__global__ __launch_bounds__(256) void aux_kernel(
    const float* __restrict__ probs, const int* __restrict__ counters,
    float* __restrict__ aux_out)
{
    __shared__ float red[256];
    __shared__ float auxs;
    if (threadIdx.x == 0) auxs = 0.f;
    __syncthreads();
    for (int e = 0; e < EE; ++e) {
        float s = 0.f;
        for (int i = threadIdx.x; i < TT; i += 256) s += probs[(size_t)i * EE + e];
        red[threadIdx.x] = s;
        __syncthreads();
        for (int w = 128; w > 0; w >>= 1) {
            if (threadIdx.x < w) red[threadIdx.x] += red[threadIdx.x + w];
            __syncthreads();
        }
        if (threadIdx.x == 0) {
            float P = red[0] / (float)TT;               // mean routing prob
            float f = (float)counters[e] / (float)TKP;  // dispatch fraction
            auxs += f * P;
        }
        __syncthreads();
    }
    if (threadIdx.x == 0) *aux_out = 0.01f * (float)EE * auxs;
}

// ---------------- launch ----------------
extern "C" void kernel_launch(void* const* d_in, const int* in_sizes, int n_in,
                              void* d_out, int out_size, void* d_ws, size_t ws_size,
                              hipStream_t stream) {
    const float* x      = (const float*)d_in[0];
    const float* gate_w = (const float*)d_in[1];
    const float* w_gate = (const float*)d_in[2];
    const float* w_up   = (const float*)d_in[3];
    const float* w_down = (const float*)d_in[4];
    float* out = (float*)d_out;

    char* ws = (char*)d_ws;
    int*   counters = (int*)(ws);                                  // 64 ints
    float* topw     = (float*)(ws + 1024);                         // 8192 f
    int*   dest     = (int*)(ws + 64 * 1024);                      // 8192 i
    float* probs    = (float*)(ws + (size_t)1 * 1024 * 1024);      // 1 MB
    bf16*  xb       = (bf16*)(ws + (size_t)2  * 1024 * 1024);      // 32 MB
    bf16*  hbuf     = (bf16*)(ws + (size_t)34 * 1024 * 1024);      // 32 MB
    float* ybuf     = (float*)(ws + (size_t)66 * 1024 * 1024);     // 64 MB

    init_counters<<<1, 64, 0, stream>>>(counters);
    router_kernel<<<TT, 64, 0, stream>>>(x, gate_w, probs, topw, dest, counters, xb);
    expert_gemm<true ><<<dim3(FF / BN, CAP / BM, EE), 256, 0, stream>>>(xb,   w_gate, w_up, hbuf, (float*)0);
    expert_gemm<false><<<dim3(HH / BN, CAP / BM, EE), 256, 0, stream>>>(hbuf, w_down, (const float*)0, (bf16*)0, ybuf);
    combine_kernel<<<TT, 256, 0, stream>>>(dest, topw, ybuf, out);
    aux_kernel<<<1, 256, 0, stream>>>(probs, counters, out + (size_t)TT * HH);
}
